// SpatiotemporalAttention_81269371175320
// MI455X (gfx1250) — matrix operand order repaired and checked by reading, not compiled, not run on hardware
//
#include <hip/hip_runtime.h>
#include <hip/hip_bf16.h>

// ---------------------------------------------------------------------------
// CDNA5 (gfx1250) windowed spatiotemporal attention, bf16 WMMA pipeline.
//   B=1, N=65536, C=512, NH=8, HD=64, windows: 512 windows x 128 tokens.
// Kernels: weight-convert -> QKV proj -> per-(win,head) attention -> out proj.
// Matrix math: v_wmma_f32_16x16x32_bf16 (f32 accumulate).
// LDS staging of pure-bf16 tiles: global_load_async_to_lds_b128 (ASYNCcnt).
// ---------------------------------------------------------------------------

typedef __attribute__((ext_vector_type(16))) __bf16 v16bf;
typedef __attribute__((ext_vector_type(8)))  float  v8f;

union FragBF {
    v16bf v;
    unsigned short u[16];
    uint4 q[2];
};

__device__ __forceinline__ unsigned short f2bf(float x) {
    union { float f; unsigned u; } v; v.f = x;
    unsigned r = v.u + 0x7FFFu + ((v.u >> 16) & 1u);   // round-to-nearest-even
    return (unsigned short)(r >> 16);
}

__device__ __forceinline__ v8f wmma_bf16(v16bf a, v16bf b, v8f c) {
    // (neg_a, A, neg_b, B, c_mod, C, reuse_a, reuse_b)
    return __builtin_amdgcn_wmma_f32_16x16x32_bf16(false, a, false, b, (short)0, c,
                                                   false, false);
}

// Async 16B copy global -> LDS (CDNA5 GLOBAL_LOAD_ASYNC_TO_LDS_B128, ASYNCcnt).
// Low 32 bits of a generic shared pointer are the LDS byte offset on AMDGPU.
__device__ __forceinline__ void async_ld_b128(void* lds, const void* g) {
    unsigned       l = (unsigned)(size_t)lds;
    unsigned long long a = (unsigned long long)(size_t)g;
    asm volatile("global_load_async_to_lds_b128 %0, %1, off"
                 :: "v"(l), "v"(a) : "memory");
}
__device__ __forceinline__ void wait_async0() {
    asm volatile("s_wait_asynccnt 0" ::: "memory");
}

// A fragment: 16x32 bf16 tile from a row-major [*, ld] bf16 matrix.
// Lanes 0-15 hold row m, K {kb+0..7, kb+16..23}; lanes 16-31 hold
// K {kb+8..15, kb+24..31}.  Two 16B loads per lane.
__device__ __forceinline__ v16bf load_a_frag(const unsigned short* __restrict__ base,
                                             int ld, int mbase, int kbase, int lane) {
    FragBF f;
    int m   = mbase + (lane & 15);
    int klo = (lane & 16) ? 8 : 0;
    const unsigned short* p = base + m * ld + kbase + klo;
    f.q[0] = *reinterpret_cast<const uint4*>(p);
    f.q[1] = *reinterpret_cast<const uint4*>(p + 16);
    return f.v;
}

// B fragment: 32x16 with B[k][n] = W[n][k], W row-major [N, ld].
// Lane (n = lane&15): lanes 0-15 -> K {kb+0..7, kb+16..23},
// lanes 16-31 -> K {kb+8..15, kb+24..31}.  Two 16B loads per lane.
__device__ __forceinline__ v16bf load_b_frag_wt(const unsigned short* __restrict__ base,
                                                int ld, int nbase, int kbase, int lane) {
    FragBF f;
    int n  = nbase + (lane & 15);
    int k0 = kbase + ((lane & 16) ? 8 : 0);
    const unsigned short* p = base + n * ld + k0;
    f.q[0] = *reinterpret_cast<const uint4*>(p);
    f.q[1] = *reinterpret_cast<const uint4*>(p + 16);
    return f.v;
}

// ---------------------------------------------------------------------------
// Kernel 0: convert 4 x (512x512) f32 weights to bf16 in workspace.
// ---------------------------------------------------------------------------
__global__ __launch_bounds__(256) void wconv_kernel(
    const float* __restrict__ Wq, const float* __restrict__ Wk,
    const float* __restrict__ Wv, const float* __restrict__ Wo,
    unsigned short* __restrict__ out)
{
    int idx = blockIdx.x * 256 + threadIdx.x;
    if (idx >= 4 * 262144) return;
    int mat = idx >> 18, off = idx & 262143;
    const float* src = (mat == 0) ? Wq : (mat == 1) ? Wk : (mat == 2) ? Wv : Wo;
    out[idx] = f2bf(src[off]);
}

// ---------------------------------------------------------------------------
// Kernel 1: per-window QKV projection.  512 blocks x 256 threads (8 waves).
// Gather window (128x512) into LDS as bf16, then Q/K/V = xw @ W^T via WMMA.
// Outputs contiguous [win][head][128][64] bf16 tiles.
// ---------------------------------------------------------------------------
__global__ __launch_bounds__(256) void qkv_kernel(
    const float* __restrict__ x, const unsigned short* __restrict__ Wb,
    unsigned short* __restrict__ Qb, unsigned short* __restrict__ Kb,
    unsigned short* __restrict__ Vb)
{
    __shared__ __align__(16) unsigned short xs[128 * 512];   // 128 KB bf16 window
    const int win = blockIdx.x;
    const int tid = threadIdx.x;
    const int tb = win >> 6, hb = (win >> 3) & 7, wb = win & 7;

    // Gather + f32->bf16 convert, 4-wide: float4 global reads, b64 LDS stores.
    for (int v4 = tid; v4 < 128 * 128; v4 += 256) {
        int l  = v4 >> 7;
        int c4 = (v4 & 127) << 2;
        int dt = l >> 6, dh = (l >> 3) & 7, dw = l & 7;
        int n = ((tb * 2 + dt) << 12) | ((hb * 8 + dh) << 6) | (wb * 8 + dw);
        float4 d = *reinterpret_cast<const float4*>(x + (long)n * 512 + c4);
        ushort4 h;
        h.x = f2bf(d.x); h.y = f2bf(d.y); h.z = f2bf(d.z); h.w = f2bf(d.w);
        *reinterpret_cast<ushort4*>(&xs[(l << 9) + c4]) = h;
    }
    __syncthreads();

    const int wv = tid >> 5, lane = tid & 31;
    // 3 matrices x (8 M-tiles x 32 N-tiles) = 768 output tiles, round-robin per wave.
    for (int t = wv; t < 768; t += 8) {
        int mat = t >> 8;
        int rem = t & 255;
        int mt = rem >> 5, nt = rem & 31;
        const unsigned short* W = Wb + (long)mat * 262144;
        v8f acc = {};
        for (int ks = 0; ks < 16; ++ks) {
            v16bf a = load_a_frag(xs, 512, mt * 16, ks * 32, lane);
            v16bf b = load_b_frag_wt(W, 512, nt * 16, ks * 32, lane);
            acc = wmma_bf16(a, b, acc);
        }
        int head = nt >> 2;
        int hdb  = (nt * 16) & 63;
        unsigned short* dst = (mat == 0) ? Qb : (mat == 1) ? Kb : Vb;
        long base = (((long)win * 8 + head) * 128 + mt * 16) * 64 + hdb + (lane & 15);
        int mo = (lane & 16) ? 8 : 0;
#pragma unroll
        for (int v = 0; v < 8; ++v)
            dst[base + (long)(v + mo) * 64] = f2bf(acc[v]);
    }
}

// ---------------------------------------------------------------------------
// Kernel 2: attention for one (window, head).  4096 blocks x 256 threads.
// Q/K staged to LDS via async b128 copies; V staged transposed (64x128) so
// PV B-fragments are contiguous.  S = (Q K^T)*scale -> softmax (1/sum folded
// into the PV epilogue) -> O = P V.
// ---------------------------------------------------------------------------
__global__ __launch_bounds__(256) void attn_kernel(
    const unsigned short* __restrict__ Qb, const unsigned short* __restrict__ Kb,
    const unsigned short* __restrict__ Vb, unsigned short* __restrict__ Ob)
{
    __shared__ __align__(16) unsigned char smem[115200];
    unsigned short* Qs  = (unsigned short*)smem;              // 16 KB  (128x64)
    unsigned short* Ks  = (unsigned short*)(smem + 16384);    // 16 KB  (128x64)
    unsigned short* Vt  = (unsigned short*)(smem + 32768);    // 16 KB  (64x128, V^T)
    float*          Ss  = (float*)(smem + 49152);             // 64 KB  (128x128 f32)
    float*          rs  = (float*)(smem + 114688);            // 512 B  row 1/sum
    unsigned short* Ps  = (unsigned short*)smem;              // 32 KB, aliases dead Q+K

    const int wh  = blockIdx.x;           // win*8 + head
    const int tid = threadIdx.x;
    const long qkv0 = (long)wh * (128 * 64);

    // Q, K: straight bf16 tile copies -> async global->LDS b128 (ASYNCcnt).
    for (int i = tid; i < 1024; i += 256) {            // 1024 x 16B = 16 KB
        async_ld_b128(&Qs[i * 8], Qb + qkv0 + i * 8);
        async_ld_b128(&Ks[i * 8], Kb + qkv0 + i * 8);
    }
    // V: transpose during staging (coalesced uint4 reads, strided u16 writes).
    for (int v8 = tid; v8 < 1024; v8 += 256) {         // 8 bf16 per iteration
        int l   = v8 >> 3;
        int hd0 = (v8 & 7) << 3;
        uint4 d = *reinterpret_cast<const uint4*>(Vb + qkv0 + (l << 6) + hd0);
        const unsigned short* e = reinterpret_cast<const unsigned short*>(&d);
#pragma unroll
        for (int j = 0; j < 8; ++j) Vt[(hd0 + j) * 128 + l] = e[j];
    }
    wait_async0();
    __syncthreads();

    const int wv = tid >> 5, lane = tid & 31;
    // S = Q @ K^T (K row-major L x HD => K^T loads are contiguous W-style)
    for (int t = wv; t < 64; t += 8) {
        int mt = t >> 3, nt = t & 7;
        v8f acc = {};
        for (int ks = 0; ks < 2; ++ks) {
            v16bf a = load_a_frag(Qs, 64, mt * 16, ks * 32, lane);
            v16bf b = load_b_frag_wt(Ks, 64, nt * 16, ks * 32, lane);
            acc = wmma_bf16(a, b, acc);
        }
        int nn = nt * 16 + (lane & 15);
        int mb = mt * 16 + ((lane & 16) ? 8 : 0);
#pragma unroll
        for (int v = 0; v < 8; ++v)
            Ss[(mb + v) * 128 + nn] = acc[v] * 0.125f;   // * 1/sqrt(64)
    }
    __syncthreads();

    // Row softmax: thread r owns row r.  Store un-normalized exp() as bf16 P.
    if (tid < 128) {
        float mx = -3.0e38f;
        for (int j = 0; j < 128; ++j) mx = fmaxf(mx, Ss[tid * 128 + j]);
        float s = 0.0f;
        for (int j = 0; j < 128; ++j) {
            float e = __expf(Ss[tid * 128 + j] - mx);
            s += e;
            Ps[tid * 128 + j] = f2bf(e);
        }
        rs[tid] = 1.0f / s;
    }
    __syncthreads();

    // O = P @ V, scaled per-row by 1/sum, written bf16 to [win][l][head*64+hd].
    // B-frags come from V^T (64x128) via the contiguous W-style loader.
    const int head = wh & 7;
    const int win  = wh >> 3;
    for (int t = wv; t < 32; t += 8) {
        int mt = t >> 2, nt = t & 3;
        v8f acc = {};
        for (int ks = 0; ks < 4; ++ks) {
            v16bf a = load_a_frag(Ps, 128, mt * 16, ks * 32, lane);
            v16bf b = load_b_frag_wt(Vt, 128, nt * 16, ks * 32, lane);
            acc = wmma_bf16(a, b, acc);
        }
        int nn = nt * 16 + (lane & 15);
        int mb = mt * 16 + ((lane & 16) ? 8 : 0);
#pragma unroll
        for (int v = 0; v < 8; ++v) {
            float o = acc[v] * rs[mb + v];
            Ob[(((long)win * 128) + mb + v) * 512 + head * 64 + nn] = f2bf(o);
        }
    }
}

// ---------------------------------------------------------------------------
// Kernel 3: output projection + window merge.  512 blocks x 256 threads.
// O window staged via async b128 copies; y = O @ Wo^T + bo scattered back to
// the original [N, C] token layout (f32).
// ---------------------------------------------------------------------------
__global__ __launch_bounds__(256) void oproj_kernel(
    const unsigned short* __restrict__ Ob, const unsigned short* __restrict__ Wob,
    const float* __restrict__ bo, float* __restrict__ y)
{
    __shared__ __align__(16) unsigned short os[128 * 512];   // 128 KB
    const int win = blockIdx.x;
    const int tid = threadIdx.x;
    const long o0 = (long)win * (128 * 512);
    for (int i = tid; i < 8192; i += 256)                    // 8192 x 16B = 128 KB
        async_ld_b128(&os[i * 8], Ob + o0 + i * 8);
    wait_async0();
    __syncthreads();

    const int tb = win >> 6, hb = (win >> 3) & 7, wb = win & 7;
    const int wv = tid >> 5, lane = tid & 31;
    for (int t = wv; t < 256; t += 8) {
        int mt = t >> 5, nt = t & 31;
        v8f acc = {};
        for (int ks = 0; ks < 16; ++ks) {
            v16bf a = load_a_frag(os, 512, mt * 16, ks * 32, lane);
            v16bf b = load_b_frag_wt(Wob, 512, nt * 16, ks * 32, lane);
            acc = wmma_bf16(a, b, acc);
        }
        int nn = nt * 16 + (lane & 15);
        float bias = bo[nn];
        int mo = (lane & 16) ? 8 : 0;
#pragma unroll
        for (int v = 0; v < 8; ++v) {
            int l = mt * 16 + mo + v;
            int dt = l >> 6, dh = (l >> 3) & 7, dw = l & 7;
            int n = ((tb * 2 + dt) << 12) | ((hb * 8 + dh) << 6) | (wb * 8 + dw);
            y[(long)n * 512 + nn] = acc[v] + bias;
        }
    }
}

// ---------------------------------------------------------------------------
// Launch.  Workspace layout (ushorts):
//   [0 .. 1M)  : 4 bf16 weight matrices (Wq,Wk,Wv,Wo), 262144 each
//   Qb/Kb/Vb   : 512*8*128*64 = 33554432 each
//   Ob         : 512*128*512  = 33554432
// Total ~258 MB.
// ---------------------------------------------------------------------------
extern "C" void kernel_launch(void* const* d_in, const int* in_sizes, int n_in,
                              void* d_out, int out_size, void* d_ws, size_t ws_size,
                              hipStream_t stream) {
    const float* x  = (const float*)d_in[0];
    const float* Wq = (const float*)d_in[1];
    const float* Wk = (const float*)d_in[2];
    const float* Wv = (const float*)d_in[3];
    const float* Wo = (const float*)d_in[4];
    const float* bo = (const float*)d_in[5];
    float* y = (float*)d_out;

    unsigned short* us = (unsigned short*)d_ws;
    unsigned short* Wb = us;                       // 4 * 262144
    unsigned short* Qb = us + 1048576;
    unsigned short* Kb = Qb + 33554432;
    unsigned short* Vb = Kb + 33554432;
    unsigned short* Ob = Vb + 33554432;

    wconv_kernel<<<4096, 256, 0, stream>>>(Wq, Wk, Wv, Wo, Wb);
    qkv_kernel<<<512, 256, 0, stream>>>(x, Wb, Qb, Kb, Vb);
    attn_kernel<<<4096, 256, 0, stream>>>(Qb, Kb, Vb, Ob);
    oproj_kernel<<<512, 256, 0, stream>>>(Ob, Wb + 3 * 262144, bo, y);
}
